// WavetableSynth_5282809774420
// MI455X (gfx1250) — matrix operand order repaired and checked by reading, not compiled, not run on hardware
//
#include <hip/hip_runtime.h>
#include <hip/hip_bf16.h>
#include <math.h>

// ---------------------------------------------------------------------------
// WavetableSynth for MI455X (gfx1250, wave32).
// out[b,t] = env * sum_n ts[n,t] * G[n,t],  G = tables_norm(8x512) @ P(512xT)
// where P[s,t] = Gaussian(rng[s]; phase[t], sigma[t]).
// K=512 reduction on v_wmma_f32_16x16x32_bf16; P generated in-register
// (v_exp_f32); A (tables) pre-staged in WMMA fragment layout and copied
// into LDS via gfx1250 async global->LDS loads (ASYNCcnt path).
// ---------------------------------------------------------------------------

typedef __attribute__((ext_vector_type(16))) __bf16 v16bf;
typedef __attribute__((ext_vector_type(8)))  float  v8f;
typedef __attribute__((ext_vector_type(4)))  int    v4i;

#define N_BATCH    4
#define N_FRAMES   128
#define N_TABLES   8
#define TABLE_SIZE 512
#define T_TOTAL    32768
#define UPS        256                  // T_TOTAL / N_FRAMES
#define RNG_STEP   (512.0f / 511.0f)    // linspace(0, 512, 512) step
#define MIN_FREQ   (40.0f / 11025.0f)
#define MAX_FREQ   (4000.0f / 11025.0f)
#define MAX_SMOOTH 0.05f

#if defined(__has_builtin)
#if __has_builtin(__builtin_amdgcn_global_load_async_to_lds_b128)
#define WTS_HAVE_ASYNC_LDS 1
#endif
#if __has_builtin(__builtin_amdgcn_s_wait_asynccnt)
#define WTS_HAVE_WAIT_ASYNC 1
#endif
#endif

typedef __attribute__((address_space(1))) v4i* gv4i_ptr;   // global
typedef __attribute__((address_space(3))) v4i* sv4i_ptr;   // LDS

__device__ __forceinline__ float interp_frame(const float* fr, int t) {
    // torch F.interpolate(mode='linear', align_corners=False), 128 -> 32768
    float src = ((float)t + 0.5f) * (1.0f / (float)UPS) - 0.5f;
    src = fminf(fmaxf(src, 0.0f), (float)(N_FRAMES - 1));
    int i0 = (int)src;                       // src >= 0 -> trunc == floor
    int i1 = i0 + 1; if (i1 > N_FRAMES - 1) i1 = N_FRAMES - 1;
    float w = src - (float)i0;
    return fr[i0] * (1.0f - w) + fr[i1] * w;
}

__device__ __forceinline__ unsigned short f32_to_bf16_bits(float x) {
    __bf16 h = (__bf16)x;
    unsigned short u; __builtin_memcpy(&u, &h, 2);
    return u;
}

// ---------------------------------------------------------------------------
// Kernel 1: tiny prep. Frame-rate transforms + bf16 A-fragment staging.
// A-fragment layout per CDNA5 ISA 7.12.2, 16-bit A 16x32 (MxK):
//   lane L: M = L%16 ; VGPR v, half=(L/16): K = (v<4?0:16) + (v&3)*2 + half*8 (+0/1)
// Stored as afrag[kk][lane][e] (e = vector element, pairs packed per dword).
// ---------------------------------------------------------------------------
__global__ void wts_prep_kernel(const float* __restrict__ env,
                                const float* __restrict__ tsel,
                                const float* __restrict__ freq,
                                const float* __restrict__ smooth,
                                const float* __restrict__ wavetables,
                                float* __restrict__ envfr,
                                float* __restrict__ ffr,
                                float* __restrict__ sfr,
                                float* __restrict__ tsfr,
                                unsigned short* __restrict__ afrag) {
    __shared__ float invmax[N_TABLES];
    int tid = threadIdx.x;

    if (tid < N_TABLES) {
        float m = 0.0f;
        for (int k = 0; k < TABLE_SIZE; ++k)
            m = fmaxf(m, fabsf(wavetables[tid * TABLE_SIZE + k]));
        invmax[tid] = 1.0f / (m + 1e-8f);
    }
    __syncthreads();

    // 16 K-chunks * 32 lanes * 16 elems = 8192 bf16 entries (16 KB)
    for (int g = tid; g < 16 * 32 * 16; g += blockDim.x) {
        int kk   = g >> 9;
        int rem  = g & 511;
        int lane = rem >> 4;
        int e    = rem & 15;
        int m    = lane & 15;
        int half = lane >> 4;
        int v    = e >> 1;
        int kBase = ((v < 4) ? 0 : 16) + (v & 3) * 2 + half * 8;
        int k     = kk * 32 + kBase + (e & 1);
        float val = (m < N_TABLES) ? wavetables[m * TABLE_SIZE + k] * invmax[m] : 0.0f;
        afrag[g] = f32_to_bf16_bits(val);
    }

    // frame-rate controls: |env|, sigmoid freq/smoothness, softmax table mix
    if (tid < N_BATCH * N_FRAMES) {
        int b = tid / N_FRAMES, fr = tid % N_FRAMES;
        envfr[tid] = fabsf(env[tid]);
        ffr[tid] = MIN_FREQ + (MAX_FREQ - MIN_FREQ) / (1.0f + __expf(-freq[tid]));
        sfr[tid] = MAX_SMOOTH / (1.0f + __expf(-smooth[tid]));

        float vals[N_TABLES];
        float mx = -1e30f;
        for (int n = 0; n < N_TABLES; ++n) {
            vals[n] = tsel[(b * N_TABLES + n) * N_FRAMES + fr];
            mx = fmaxf(mx, vals[n]);
        }
        float sum = 0.0f;
        for (int n = 0; n < N_TABLES; ++n) { vals[n] = __expf(vals[n] - mx); sum += vals[n]; }
        float inv = 1.0f / sum;
        for (int n = 0; n < N_TABLES; ++n)
            tsfr[(b * N_TABLES + n) * N_FRAMES + fr] = vals[n] * inv;
    }
}

// ---------------------------------------------------------------------------
// Kernel 2: phase = cumsum(interp(freq)) mod 1.0 — hierarchical f32 scan,
// one block per batch, 256 threads * 128 contiguous samples each.
// ---------------------------------------------------------------------------
__global__ void wts_phase_scan_kernel(const float* __restrict__ ffr,
                                      float* __restrict__ phase) {
    int b = blockIdx.x, tid = threadIdx.x;
    __shared__ float ffrS[N_FRAMES];
    __shared__ float part[256];
    if (tid < N_FRAMES) ffrS[tid] = ffr[b * N_FRAMES + tid];
    __syncthreads();

    const int PER = T_TOTAL / 256;     // 128
    int base = tid * PER;
    float local = 0.0f;
    for (int j = 0; j < PER; ++j) local += interp_frame(ffrS, base + j);
    part[tid] = local;
    __syncthreads();
    if (tid == 0) {
        float run = 0.0f;
        for (int i = 0; i < 256; ++i) { float v = part[i]; part[i] = run; run += v; }
    }
    __syncthreads();
    float run = part[tid];
    for (int j = 0; j < PER; ++j) {
        run += interp_frame(ffrS, base + j);
        phase[b * T_TOTAL + base + j] = run - floorf(run);   // x % 1.0, x >= 0
    }
}

// ---------------------------------------------------------------------------
// Kernel 3: main synth. One wave32 = one 16-sample tile of one batch.
// G(16x16 f32) += A(16x32 bf16, tables) x B(32x16 bf16, Gaussian read),
// 16 K-chunks cover s = 0..511.
// B fragment layout (dense 16-bit B, K x N): lane L holds column N = L%16,
// K = (L/16)*8 + e for e < 8, 16 + (L/16)*8 + (e-8) for e >= 8 (same order as A).
// ---------------------------------------------------------------------------
__global__ void __launch_bounds__(256)
wts_synth_kernel(const float* __restrict__ envfr,
                 const float* __restrict__ sfr,
                 const float* __restrict__ tsfr,
                 const float* __restrict__ phase,
                 const unsigned short* __restrict__ afrag,
                 float* __restrict__ out) {
    __shared__ unsigned short afragS[16 * 32 * 16];   // 16 KB, A fragments

    int tid = threadIdx.x;
#if WTS_HAVE_ASYNC_LDS
    {   // gfx1250 async global->LDS copy: no VGPR round-trip, ASYNCcnt-tracked
        v4i* srcg = (v4i*)afrag;            // 16-byte chunks
        v4i* dsts = (v4i*)afragS;
        for (int i = tid; i < (16 * 32 * 16) / 8; i += blockDim.x) {
            __builtin_amdgcn_global_load_async_to_lds_b128(
                (gv4i_ptr)(srcg + i), (sv4i_ptr)(dsts + i),
                /*offset=*/0, /*cpol=*/0);
        }
#if WTS_HAVE_WAIT_ASYNC
        __builtin_amdgcn_s_wait_asynccnt(0);
#else
        asm volatile("s_wait_asynccnt 0x0" ::: "memory");
#endif
    }
#else
    {   // fallback: block-stage A fragments via VGPRs (128-bit copies)
        const uint4* src = (const uint4*)afrag;
        uint4* dst = (uint4*)afragS;
        for (int i = tid; i < (16 * 32 * 16) / 8; i += blockDim.x) dst[i] = src[i];
    }
#endif
    __syncthreads();

    int wid  = tid >> 5;
    int lane = tid & 31;
    int tile = blockIdx.x * 8 + wid;          // 8 waves/block, 8192 tiles total
    int b    = tile >> 11;                    // 2048 tiles per batch
    int t0   = (tile & 2047) << 4;
    int c    = lane & 15;                     // this lane's output column
    int t    = t0 + c;

    float envv = interp_frame(envfr + b * N_FRAMES, t);
    float sv   = interp_frame(sfr   + b * N_FRAMES, t);
    float fv   = phase[b * T_TOTAL + t];
    float tsv[N_TABLES];
    #pragma unroll
    for (int n = 0; n < N_TABLES; ++n)
        tsv[n] = interp_frame(tsfr + (b * N_TABLES + n) * N_FRAMES, t);

    // read[s] = exp(-(rng[s]-f)^2/(2 s^2)) / (s*sqrt(2*pi))
    float amp    = 1.0f / (sv * 2.5066282746f);
    float ninv2v = -0.5f / (sv * sv);
    int   krow   = (lane >> 4) << 3;          // 0 or 8: lane-half K offset in B

    v8f acc = {};
    #pragma unroll
    for (int kk = 0; kk < 16; ++kk) {
        v16bf a;
        __builtin_memcpy(&a, &afragS[(kk * 32 + lane) * 16], 32);

        v16bf bb;
        int kbase = kk * 32 + krow;
        #pragma unroll
        for (int e = 0; e < 16; ++e) {
            float r = (float)(kbase + e + ((e >> 3) << 3)) * RNG_STEP;
            float d = r - fv;
            bb[e] = (__bf16)(amp * __expf(ninv2v * d * d));
        }
        acc = __builtin_amdgcn_wmma_f32_16x16x32_bf16(
            /*neg_a=*/false, a, /*neg_b=*/false, bb,
            /*c_mod=*/(short)0, acc, /*reuse_a=*/false, /*reuse_b=*/false);
    }

    // D rows 0..7 (valid tables) live in VGPRs 0..7 of lanes 0..15
    if (lane < 16) {
        float r = 0.0f;
        #pragma unroll
        for (int n = 0; n < N_TABLES; ++n) r += tsv[n] * acc[n];
        out[b * T_TOTAL + t] = r * envv;
    }
}

// ---------------------------------------------------------------------------
// Launch
// ---------------------------------------------------------------------------
extern "C" void kernel_launch(void* const* d_in, const int* in_sizes, int n_in,
                              void* d_out, int out_size, void* d_ws, size_t ws_size,
                              hipStream_t stream) {
    const float* env        = (const float*)d_in[0];   // (4,1,128)
    const float* tsel       = (const float*)d_in[1];   // (4,8,128)
    const float* freq       = (const float*)d_in[2];   // (4,1,128)
    const float* smooth     = (const float*)d_in[3];   // (4,1,128)
    const float* wavetables = (const float*)d_in[4];   // (8,512)
    float* out = (float*)d_out;                        // (4,1,32768)

    float* wsf   = (float*)d_ws;
    float* phase = wsf;                                   // 4*32768
    float* envfr = phase + N_BATCH * T_TOTAL;             // 512
    float* ffr   = envfr + N_BATCH * N_FRAMES;            // 512
    float* sfr   = ffr   + N_BATCH * N_FRAMES;            // 512
    float* tsfr  = sfr   + N_BATCH * N_FRAMES;            // 4096
    unsigned short* afrag = (unsigned short*)(tsfr + N_BATCH * N_TABLES * N_FRAMES); // 8192 bf16

    wts_prep_kernel<<<1, 512, 0, stream>>>(env, tsel, freq, smooth, wavetables,
                                           envfr, ffr, sfr, tsfr, afrag);
    wts_phase_scan_kernel<<<N_BATCH, 256, 0, stream>>>(ffr, phase);

    int tiles = N_BATCH * (T_TOTAL / 16);   // 8192
    wts_synth_kernel<<<tiles / 8, 256, 0, stream>>>(envfr, sfr, tsfr, phase, afrag, out);
}
